// TrajsEncoder_74526272520513
// MI455X (gfx1250) — matrix operand order repaired and checked by reading, not compiled, not run on hardware
//
#include <hip/hip_runtime.h>
#include <math.h>

#define SSW 320          // stride of scale/shift & stats buffers (max padded width)
#define EPSV 1e-5f

typedef __attribute__((ext_vector_type(2)))  _Float16 v2h;
typedef __attribute__((ext_vector_type(8)))  _Float16 v8h;
typedef __attribute__((ext_vector_type(16))) _Float16 v16h;
typedef __attribute__((ext_vector_type(8)))  float    v8f;

// ------------------------------- device helpers -----------------------------
__device__ __forceinline__ float actf(float v, int mode){
  if (mode == 1) return v > 0.f ? v : 0.2f * v;   // LeakyReLU(0.2)
  if (mode == 2) return tanhf(v);
  return v;                                        // identity
}
__device__ __forceinline__ void atomicMaxF(float* a, float v){
  int vi = __float_as_int(v);
  if (vi >= 0) atomicMax((int*)a, vi);
  else         atomicMin((unsigned int*)a, (unsigned int)vi);
}

// ------------------------------- utility kernels ----------------------------
__global__ void k_fill(float* p, int n, float v){
  int i = blockIdx.x * blockDim.x + threadIdx.x;
  if (i < n) p[i] = v;
}
__global__ void k_to_half(const float* __restrict__ in, int M, int cols, int pad,
                          _Float16* __restrict__ out){
  int i = blockIdx.x * blockDim.x + threadIdx.x;
  if (i >= M * pad) return;
  int r = i / pad, c = i - r * pad;
  out[i] = (_Float16)(c < cols ? in[(size_t)r * cols + c] : 0.f);
}
// pad weights [N][K] f32 -> [padN][padK] f16 (zero pad), pad bias to f32[padN]
__global__ void k_prep_w(const float* __restrict__ W, const float* __restrict__ b,
                         int K, int N, int padK, int padN,
                         _Float16* __restrict__ Wp, float* __restrict__ bp){
  int i = blockIdx.x * blockDim.x + threadIdx.x;
  if (i < padK * padN){
    int n = i / padK, k = i - n * padK;
    Wp[i] = (_Float16)((n < N && k < K) ? W[(size_t)n * K + k] : 0.f);
  }
  if (i < padN) bp[i] = (i < N) ? b[i] : 0.f;
}
// message-MLP layer0 weights: real cols = [e(8) | x_src(xd) | x_dst(xd)]
// padded cols = [32 | xpad | xpad]
__global__ void k_prep_w_seg(const float* __restrict__ W, const float* __restrict__ b,
                             int K, int N, int padK, int padN, int xd, int xpad,
                             _Float16* __restrict__ Wp, float* __restrict__ bp){
  int i = blockIdx.x * blockDim.x + threadIdx.x;
  if (i < padK * padN){
    int n = i / padK, c = i - n * padK;
    int rc = -1;
    if (c < 32){ if (c < 8) rc = c; }
    else if (c < 32 + xpad){ int c2 = c - 32;        if (c2 < xd) rc = 8 + c2; }
    else                   { int c2 = c - 32 - xpad; if (c2 < xd) rc = 8 + xd + c2; }
    Wp[i] = (_Float16)((rc >= 0 && n < N) ? W[(size_t)n * K + rc] : 0.f);
  }
  if (i < padN) bp[i] = (i < N) ? b[i] : 0.f;
}
// column sums & sums-of-squares (training-mode BatchNorm), vectorized by 2
__global__ void k_colstats(const _Float16* __restrict__ Y, int M, int ld, int cols,
                           float* __restrict__ stats){
  extern __shared__ char smemraw[];
  float* s = (float*)smemraw;                       // [2*cols]
  for (int i = threadIdx.x; i < 2 * cols; i += blockDim.x) s[i] = 0.f;
  __syncthreads();
  int r0 = blockIdx.x * 512;
  int rows = min(512, M - r0);
  const int cols2 = cols >> 1;
  int tot2 = rows * cols2;
  int r = (int)threadIdx.x / cols2, c2 = (int)threadIdx.x - r * cols2;
  for (int i = threadIdx.x; i < tot2; i += blockDim.x){
    v2h a = *(const v2h*)(Y + (size_t)(r0 + r) * ld + 2 * c2);
    float v0 = (float)a[0], v1 = (float)a[1];
    int c = 2 * c2;
    atomicAdd(&s[c],            v0);
    atomicAdd(&s[cols + c],     v0 * v0);
    atomicAdd(&s[c + 1],        v1);
    atomicAdd(&s[cols + c + 1], v1 * v1);
    c2 += blockDim.x; while (c2 >= cols2){ c2 -= cols2; ++r; }
  }
  __syncthreads();
  for (int i = threadIdx.x; i < cols; i += blockDim.x){
    atomicAdd(&stats[i],       s[i]);
    atomicAdd(&stats[SSW + i], s[cols + i]);
  }
}
// stats -> per-column affine:  y_norm = scale*y + shift  (scale=rstd*gamma)
__global__ void k_finalize(const float* __restrict__ stats, int M, int N, int cols,
                           const float* __restrict__ gamma, const float* __restrict__ beta,
                           float* __restrict__ ss){
  int c = blockIdx.x * blockDim.x + threadIdx.x;
  if (c >= cols) return;
  float mean = stats[c] / (float)M;
  float var  = stats[SSW + c] / (float)M - mean * mean;
  float rstd = rsqrtf(fmaxf(var, 0.f) + EPSV);
  float g  = (c < N) ? gamma[c] : 0.f;
  float bt = (c < N) ? beta[c]  : 0.f;
  float sc = rstd * g;
  ss[c] = sc;
  ss[SSW + c] = bt - mean * sc;
}

// ------------------------------- WMMA GEMM ----------------------------------
// Y[M][padN](f16) = act(inSS(A)) @ Wp^T + b.   A tile + W staged in LDS.
// gatherXW != 0: A row = edge; columns [0,32)=E1raw(tanh∘eSS),
//                [32,32+xw)=Xraw[src], [32+xw,..)=Xraw[dst] (tanh∘xSS).
__global__ __launch_bounds__(256) void k_gemm(
    const _Float16* __restrict__ A, int lda,
    const float* __restrict__ inSS, int inAct,
    const _Float16* __restrict__ Wp, const float* __restrict__ bp,
    int padK, int padN,
    _Float16* __restrict__ Y, int ldy, int Mrows,
    int gatherXW,
    const _Float16* __restrict__ Eraw, const float* __restrict__ eSS,
    const _Float16* __restrict__ Xraw, const float* __restrict__ xSS,
    const int* __restrict__ eidx)
{
  extern __shared__ char smemraw[];
  _Float16* Wl = (_Float16*)smemraw;                 // [padN][padK]
  _Float16* Al = Wl + (size_t)padN * padK;           // [128][padK]
  const int tid = threadIdx.x;
  // stage weights: 16-byte chunks (padN*padK is a multiple of 1024 halves)
  {
    const uint4* Ws = (const uint4*)Wp;
    uint4* Wd = (uint4*)Wl;
    int nchunks = (padN * padK) >> 3;
    for (int i = tid; i < nchunks; i += 256) Wd[i] = Ws[i];
  }
  const int rowBase = blockIdx.x * 128;
  const int rows = min(128, Mrows - rowBase);        // always a multiple of 16
  const int cols2 = padK >> 1;
  const int tot2 = rows * cols2;                     // pairs of halves
  if (gatherXW == 0){
    int r = tid / cols2, c2 = tid - (tid / cols2) * cols2;
    for (int i = tid; i < tot2; i += 256){
      int c = 2 * c2;
      v2h a = *(const v2h*)(A + (size_t)(rowBase + r) * lda + c);
      v2h o;
      o[0] = (_Float16)actf((float)a[0] * inSS[c]     + inSS[SSW + c],     inAct);
      o[1] = (_Float16)actf((float)a[1] * inSS[c + 1] + inSS[SSW + c + 1], inAct);
      *(v2h*)(Al + (size_t)r * padK + c) = o;
      c2 += 256; while (c2 >= cols2){ c2 -= cols2; ++r; }
    }
  } else {
    const int xw = gatherXW;                         // multiple of 32: pairs never straddle
    int r = tid / cols2, c2 = tid - (tid / cols2) * cols2;
    for (int i = tid; i < tot2; i += 256){
      int c = 2 * c2;
      int e = rowBase + r;
      v2h a; const float* sp; int sc;
      if (c < 32){
        a = *(const v2h*)(Eraw + (size_t)e * 32 + c);
        sp = eSS; sc = c;
      } else {
        int cx = c - 32; int node;
        if (cx < xw){ node = eidx[e]; }
        else        { node = eidx[(size_t)Mrows + e]; cx -= xw; }
        a = *(const v2h*)(Xraw + (size_t)node * xw + cx);
        sp = xSS; sc = cx;
      }
      v2h o;
      o[0] = (_Float16)tanhf((float)a[0] * sp[sc]     + sp[SSW + sc]);
      o[1] = (_Float16)tanhf((float)a[1] * sp[sc + 1] + sp[SSW + sc + 1]);
      *(v2h*)(Al + (size_t)r * padK + c) = o;
      c2 += 256; while (c2 >= cols2){ c2 -= cols2; ++r; }
    }
  }
  __syncthreads();

  const int wave = tid >> 5, lane = tid & 31;
  const int lhalf = lane & 15, lhi = lane >> 4;
  const int nt = padN >> 4, kt = padK >> 5, rt = rows >> 4;
  for (int task = wave; task < rt * nt; task += 8){
    const int rtile = task / nt, ntile = task - (task / nt) * nt;
    const int n = ntile * 16 + lhalf;
    const float bv = bp[n];
    v8f acc;
    #pragma unroll
    for (int i = 0; i < 8; ++i) acc[i] = bv;
    // A frag: lanes 0-15 row=M, K 0..15 contiguous, lanes 16-31: K 16..31
    const _Float16* ap  = Al + (size_t)(rtile * 16 + lhalf) * padK + lhi * 16;
    // B frag: B[k][n]=W[n][k]; lanes 0-15: K 0..15 contiguous, lanes 16-31: K 16..31
    const _Float16* bpW = Wl + (size_t)n * padK + lhi * 16;
    for (int k = 0; k < kt; ++k){
      v8h alo = *(const v8h*)(ap + k * 32);
      v8h ahi = *(const v8h*)(ap + k * 32 + 8);
      v16h av = __builtin_shufflevector(alo, ahi, 0,1,2,3,4,5,6,7,8,9,10,11,12,13,14,15);
      v16h bvv = *(const v16h*)(bpW + k * 32);
      acc = __builtin_amdgcn_wmma_f32_16x16x32_f16(false, av, false, bvv,
                                                   (short)0, acc, false, false);
    }
    const int mbase = rowBase + rtile * 16 + lhi * 8;
    #pragma unroll
    for (int i = 0; i < 8; ++i)
      Y[(size_t)(mbase + i) * ldy + n] = (_Float16)acc[i];
  }
}

// ------------------------------- aggregation --------------------------------
// vectorized by 2 over the 64 message columns
__global__ void k_scatter_mean(const _Float16* __restrict__ msg, const float* __restrict__ ss,
                               const int* __restrict__ dst, float* __restrict__ agg,
                               float* __restrict__ cnt, int E){
  int i = blockIdx.x * blockDim.x + threadIdx.x;
  if (i >= E * 32) return;
  int e = i >> 5, c = (i & 31) * 2;
  v2h a = *(const v2h*)(msg + (size_t)e * 64 + c);
  float v0 = (float)a[0] * ss[c]     + ss[SSW + c];
  float v1 = (float)a[1] * ss[c + 1] + ss[SSW + c + 1];
  int d = dst[e];
  atomicAdd(&agg[(size_t)d * 64 + c],     v0);
  atomicAdd(&agg[(size_t)d * 64 + c + 1], v1);
  if (c == 0) atomicAdd(&cnt[d], 1.f);
}
__global__ void k_scatter_max(const _Float16* __restrict__ msg, const float* __restrict__ ss,
                              const int* __restrict__ dst, float* __restrict__ agg, int E){
  int i = blockIdx.x * blockDim.x + threadIdx.x;
  if (i >= E * 32) return;
  int e = i >> 5, c = (i & 31) * 2;
  v2h a = *(const v2h*)(msg + (size_t)e * 64 + c);
  float v0 = (float)a[0] * ss[c]     + ss[SSW + c];
  float v1 = (float)a[1] * ss[c + 1] + ss[SSW + c + 1];
  int d = dst[e];
  atomicMaxF(&agg[(size_t)d * 64 + c],     v0);
  atomicMaxF(&agg[(size_t)d * 64 + c + 1], v1);
}
__global__ void k_aggfin(const float* __restrict__ agg, const float* __restrict__ cnt,
                         int isMax, _Float16* __restrict__ out, int Nn){
  int i = blockIdx.x * blockDim.x + threadIdx.x;
  if (i >= Nn * 64) return;
  int r = i >> 6;
  float v = agg[i];
  if (isMax){ if (!(v * 0.f == 0.f)) v = 0.f; }       // -inf/NaN (empty seg) -> 0
  else v = v / fmaxf(cnt[r], 1.f);
  out[i] = (_Float16)v;
}
// apply BN affine (+act) elementwise: out[r][ooff+c] = act(ss(in[r][c]))
__global__ void k_bnact(const _Float16* __restrict__ in, int ld, const float* __restrict__ ss,
                        int act, _Float16* __restrict__ out, int old, int ooff, int M, int C){
  int i = blockIdx.x * blockDim.x + threadIdx.x;
  if (i >= M * C) return;
  int r = i / C, c = i - r * C;
  float v = actf((float)in[(size_t)r * ld + c] * ss[c] + ss[SSW + c], act);
  out[(size_t)r * old + ooff + c] = (_Float16)v;
}

// ------------------------------- pooling ------------------------------------
__global__ void k_gate(const _Float16* __restrict__ Yg, int ld, const float* __restrict__ ss,
                       const int* __restrict__ batch, float* __restrict__ g,
                       float* __restrict__ gmax, int Nn){
  int n = blockIdx.x * blockDim.x + threadIdx.x;
  if (n >= Nn) return;
  float v = (float)Yg[(size_t)n * ld] * ss[0] + ss[SSW];
  g[n] = v;
  atomicMaxF(&gmax[batch[n]], v);
}
__global__ void k_poolw(const float* __restrict__ g, const int* __restrict__ batch,
                        const float* __restrict__ gmax, float* __restrict__ w,
                        float* __restrict__ wsum, int Nn){
  int n = blockIdx.x * blockDim.x + threadIdx.x;
  if (n >= Nn) return;
  float v = expf(g[n] - gmax[batch[n]]);
  w[n] = v;
  atomicAdd(&wsum[batch[n]], v);
}
__global__ void k_poolacc(const _Float16* __restrict__ H, const float* __restrict__ w,
                          const float* __restrict__ wsum, const int* __restrict__ batch,
                          float* __restrict__ pooled, int Nn){
  int i = blockIdx.x * blockDim.x + threadIdx.x;
  if (i >= Nn * 192) return;
  int r = i / 192, c = i - r * 192;
  float coef = w[r] / wsum[batch[r]];
  atomicAdd(&pooled[(size_t)batch[r] * 192 + c], coef * (float)H[(size_t)r * 192 + c]);
}
__global__ void k_out(const _Float16* __restrict__ Ym, int ld, const float* __restrict__ ss,
                      float* __restrict__ out, int G, int L){
  int i = blockIdx.x * blockDim.x + threadIdx.x;
  if (i >= G * L) return;
  int g = i / L, j = i - g * L;
  out[i] = (float)Ym[(size_t)g * ld + j] * ss[j] + ss[SSW + j];
}

// ------------------------------- host side ----------------------------------
namespace {
struct Lay {
  const float *W, *b, *gam, *bet;
  int K, N, padK, padN;
  _Float16* Wp; float* bp; float* stats; float* ss;
};
struct BNp { const float *gam, *bet; int D; float* stats; float* ss; };
static inline int pad32(int v){ return (v + 31) & ~31; }
}

extern "C" void kernel_launch(void* const* d_in, const int* in_sizes, int n_in,
                              void* d_out, int out_size, void* d_ws, size_t ws_size,
                              hipStream_t stream)
{
  (void)n_in; (void)ws_size; (void)out_size;
  const float* x     = (const float*)d_in[0];
  const float* ea    = (const float*)d_in[1];
  const int*   eidx  = (const int*)d_in[2];
  const int*   batch = (const int*)d_in[3];
  const int NN = in_sizes[0] / 8;        // 50000 nodes
  const int NE = in_sizes[1] / 8;        // 800000 edges
  const int NG = 256, LAT = 8;

  // ---- parse params in setup_inputs() dict insertion order -----------------
  int ii = 4;
  auto nf = [&]() -> const float* { return (const float*)d_in[ii++]; };
  BNp bnx[3], bne[3];
  Lay netx[3][2], nete[3][2], gl[3][3], fl[3][3], gate[3], mlpl;
  const int xdim[3] = {8, 64, 128};
  auto readL = [&](Lay& L, int K, int N){
    L.W = nf(); L.b = nf(); L.gam = nf(); L.bet = nf();
    L.K = K; L.N = N; L.padK = pad32(K); L.padN = pad32(N);
  };
  for (int c = 0; c < 3; ++c){
    bnx[c].gam = nf(); bnx[c].bet = nf(); bnx[c].D = xdim[c];
    readL(netx[c][0], xdim[c], 2 * xdim[c]);
    readL(netx[c][1], 2 * xdim[c], xdim[c]);
    bne[c].gam = nf(); bne[c].bet = nf(); bne[c].D = 8;
    readL(nete[c][0], 8, 16);
    readL(nete[c][1], 16, 8);
    int gin = 8 + 2 * xdim[c];
    readL(gl[c][0], gin, 128);
    gl[c][0].padK = 32 + 2 * pad32(xdim[c]);   // segmented pad: [e|x_src|x_dst]
    readL(gl[c][1], 128, 64);
    readL(gl[c][2], 64, 64);
    readL(fl[c][0], 64, 128);
    readL(fl[c][1], 128, 64);
    readL(fl[c][2], 64, 64);
  }
  readL(gate[0], 192, 64);
  readL(gate[1], 64, 32);
  readL(gate[2], 32, 1);
  readL(mlpl, 192, 8);

  // ---- workspace bump allocator --------------------------------------------
  size_t off = 0;
  auto alloc = [&](size_t bytes) -> void* {
    void* p = (char*)d_ws + off;
    off += (bytes + 255) & ~(size_t)255;
    return p;
  };
  // big activation buffers (f16 raw linear outputs, padded widths)
  _Float16* Xh  = (_Float16*)alloc((size_t)NN * 32 * 2);
  _Float16* EH  = (_Float16*)alloc((size_t)NE * 32 * 2);
  _Float16* G0  = (_Float16*)alloc((size_t)NE * 128 * 2);
  _Float16* G1  = (_Float16*)alloc((size_t)NE * 64 * 2);
  _Float16* G2  = (_Float16*)alloc((size_t)NE * 64 * 2);
  _Float16* E0  = G1;                     // alias: dead by the time G1 is written
  _Float16* E1  = G2;                     // alias: dead by the time G2 is written
  _Float16* Xn0 = (_Float16*)alloc((size_t)NN * 256 * 2);
  _Float16* Xn1 = (_Float16*)alloc((size_t)NN * 128 * 2);
  _Float16* F0  = (_Float16*)alloc((size_t)NN * 128 * 2);
  _Float16* F1  = (_Float16*)alloc((size_t)NN * 64 * 2);
  _Float16* F2  = (_Float16*)alloc((size_t)NN * 64 * 2);
  float*    AGG = (float*)alloc((size_t)NN * 64 * 4);
  float*    CNT = (float*)alloc((size_t)NN * 4);
  _Float16* AGGH= (_Float16*)alloc((size_t)NN * 64 * 2);
  _Float16* X12 = (_Float16*)alloc((size_t)NN * 128 * 2);
  _Float16* H   = (_Float16*)alloc((size_t)NN * 192 * 2);
  _Float16* Gt0 = (_Float16*)alloc((size_t)NN * 64 * 2);
  _Float16* Gt1 = (_Float16*)alloc((size_t)NN * 32 * 2);
  _Float16* Gt2 = (_Float16*)alloc((size_t)NN * 32 * 2);
  float*    gbuf= (float*)alloc((size_t)NN * 4);
  float*    wbuf= (float*)alloc((size_t)NN * 4);
  float*    gmax= (float*)alloc((size_t)NG * 4);
  float*    wsum= (float*)alloc((size_t)NG * 4);
  float*    pooled = (float*)alloc((size_t)NG * 192 * 4);
  _Float16* Ph  = (_Float16*)alloc((size_t)NG * 192 * 2);
  _Float16* Ym  = (_Float16*)alloc((size_t)NG * 32 * 2);
  float*    unitSS = (float*)alloc(2 * SSW * 4);

  auto allocLay = [&](Lay& L){
    L.Wp = (_Float16*)alloc((size_t)L.padK * L.padN * 2);
    L.bp = (float*)alloc((size_t)L.padN * 4);
    L.stats = (float*)alloc(2 * SSW * 4);
    L.ss    = (float*)alloc(2 * SSW * 4);
  };
  auto allocBN = [&](BNp& B){
    B.stats = (float*)alloc(2 * SSW * 4);
    B.ss    = (float*)alloc(2 * SSW * 4);
  };
  for (int c = 0; c < 3; ++c){
    allocBN(bnx[c]); allocBN(bne[c]);
    for (int l = 0; l < 2; ++l){ allocLay(netx[c][l]); allocLay(nete[c][l]); }
    for (int l = 0; l < 3; ++l){ allocLay(gl[c][l]); allocLay(fl[c][l]); }
  }
  for (int l = 0; l < 3; ++l) allocLay(gate[l]);
  allocLay(mlpl);

  // ---- launch helpers -------------------------------------------------------
  auto fill = [&](float* p, int n, float v){
    k_fill<<<(n + 255) / 256, 256, 0, stream>>>(p, n, v);
  };
  auto prepL = [&](Lay& L){
    int tot = L.padK * L.padN;
    k_prep_w<<<(tot + 255) / 256, 256, 0, stream>>>(L.W, L.b, L.K, L.N, L.padK, L.padN, L.Wp, L.bp);
  };
  auto prepSeg = [&](Lay& L, int xd){
    int tot = L.padK * L.padN;
    k_prep_w_seg<<<(tot + 255) / 256, 256, 0, stream>>>(L.W, L.b, L.K, L.N, L.padK, L.padN,
                                                        xd, pad32(xd), L.Wp, L.bp);
  };
  auto runStats = [&](const _Float16* Y, int M, int ld, int cols,
                      float* stats, float* ss, const float* gam, const float* bet, int Nreal){
    fill(stats, 2 * SSW, 0.f);
    k_colstats<<<(M + 511) / 512, 256, 2 * cols * sizeof(float), stream>>>(Y, M, ld, cols, stats);
    k_finalize<<<(cols + 255) / 256, 256, 0, stream>>>(stats, M, Nreal, cols, gam, bet, ss);
  };
  auto statsOf = [&](Lay& L, const _Float16* Y, int M){
    runStats(Y, M, L.padN, L.padN, L.stats, L.ss, L.gam, L.bet, L.N);
  };
  auto runGemm = [&](const _Float16* A, int lda, const float* inSS, int inAct,
                     const Lay& L, _Float16* Y, int M){
    size_t lds = (size_t)(L.padN + 128) * L.padK * sizeof(_Float16);
    k_gemm<<<(M + 127) / 128, 256, lds, stream>>>(A, lda, inSS, inAct, L.Wp, L.bp,
        L.padK, L.padN, Y, L.padN, M, 0,
        (const _Float16*)nullptr, (const float*)nullptr,
        (const _Float16*)nullptr, (const float*)nullptr, (const int*)nullptr);
  };
  auto runGemmGather = [&](const Lay& L, const _Float16* Er, const float* eSS,
                           const _Float16* Xr, int xw, const float* xSS, _Float16* Y){
    size_t lds = (size_t)(L.padN + 128) * L.padK * sizeof(_Float16);
    k_gemm<<<(NE + 127) / 128, 256, lds, stream>>>((const _Float16*)nullptr, 0,
        (const float*)nullptr, 0, L.Wp, L.bp,
        L.padK, L.padN, Y, L.padN, NE, xw, Er, eSS, Xr, xSS, eidx);
  };

  // ---- prep: weights, unit affine, input conversion ------------------------
  fill(unitSS, SSW, 1.f);
  fill(unitSS + SSW, SSW, 0.f);
  for (int c = 0; c < 3; ++c){
    for (int l = 0; l < 2; ++l){ prepL(netx[c][l]); prepL(nete[c][l]); }
    prepSeg(gl[c][0], xdim[c]);
    prepL(gl[c][1]); prepL(gl[c][2]);
    for (int l = 0; l < 3; ++l) prepL(fl[c][l]);
  }
  for (int l = 0; l < 3; ++l) prepL(gate[l]);
  prepL(mlpl);
  k_to_half<<<(NN * 32 + 255) / 256, 256, 0, stream>>>(x, NN, 8, 32, Xh);
  k_to_half<<<(NE * 32 + 255) / 256, 256, 0, stream>>>(ea, NE, 8, 32, EH);

  // ---- the three JumpsConv layers ------------------------------------------
  const _Float16* Xin[3] = { Xh, X12, X12 };
  const int XinLd[3] = { 32, 128, 128 };
  for (int c = 0; c < 3; ++c){
    const int isMax = (c == 1);
    // node transform: bn_x -> net_x (2 layers, last tanh fused at gather)
    runStats(Xin[c], NN, XinLd[c], netx[c][0].padK,
             bnx[c].stats, bnx[c].ss, bnx[c].gam, bnx[c].bet, bnx[c].D);
    runGemm(Xin[c], XinLd[c], bnx[c].ss, 0, netx[c][0], Xn0, NN);
    statsOf(netx[c][0], Xn0, NN);
    runGemm(Xn0, netx[c][0].padN, netx[c][0].ss, 1, netx[c][1], Xn1, NN);
    statsOf(netx[c][1], Xn1, NN);
    // edge transform: bn_e -> net_e (2 layers, last tanh fused at gather)
    runStats(EH, NE, 32, 32, bne[c].stats, bne[c].ss, bne[c].gam, bne[c].bet, 8);
    runGemm(EH, 32, bne[c].ss, 0, nete[c][0], E0, NE);
    statsOf(nete[c][0], E0, NE);
    runGemm(E0, 32, nete[c][0].ss, 1, nete[c][1], E1, NE);
    statsOf(nete[c][1], E1, NE);
    // message MLP g with fused gather/concat of [e, x_src, x_dst]
    runGemmGather(gl[c][0], E1, nete[c][1].ss, Xn1, netx[c][1].padN, netx[c][1].ss, G0);
    statsOf(gl[c][0], G0, NE);
    runGemm(G0, 128, gl[c][0].ss, 1, gl[c][1], G1, NE);
    statsOf(gl[c][1], G1, NE);
    runGemm(G1, 64, gl[c][1].ss, 1, gl[c][2], G2, NE);
    statsOf(gl[c][2], G2, NE);
    // aggregate messages per destination node
    fill(AGG, NN * 64, isMax ? -INFINITY : 0.f);
    fill(CNT, NN, 0.f);
    if (isMax)
      k_scatter_max<<<(NE * 32 + 255) / 256, 256, 0, stream>>>(G2, gl[c][2].ss, eidx + NE, AGG, NE);
    else
      k_scatter_mean<<<(NE * 32 + 255) / 256, 256, 0, stream>>>(G2, gl[c][2].ss, eidx + NE, AGG, CNT, NE);
    k_aggfin<<<(NN * 64 + 255) / 256, 256, 0, stream>>>(AGG, CNT, isMax, AGGH, NN);
    // post-aggregate MLP f
    runGemm(AGGH, 64, unitSS, 0, fl[c][0], F0, NN);
    statsOf(fl[c][0], F0, NN);
    runGemm(F0, 128, fl[c][0].ss, 1, fl[c][1], F1, NN);
    statsOf(fl[c][1], F1, NN);
    runGemm(F1, 64, fl[c][1].ss, 1, fl[c][2], F2, NN);
    statsOf(fl[c][2], F2, NN);
    // materialize conv output x_c = bn(F2) into X12 / H
    if (c < 2)
      k_bnact<<<(NN * 64 + 255) / 256, 256, 0, stream>>>(F2, 64, fl[c][2].ss, 0, X12, 128, c * 64, NN, 64);
    k_bnact<<<(NN * 64 + 255) / 256, 256, 0, stream>>>(F2, 64, fl[c][2].ss, 0, H, 192, c * 64, NN, 64);
  }

  // ---- global attention pooling --------------------------------------------
  runGemm(H, 192, unitSS, 0, gate[0], Gt0, NN);
  statsOf(gate[0], Gt0, NN);
  runGemm(Gt0, 64, gate[0].ss, 1, gate[1], Gt1, NN);
  statsOf(gate[1], Gt1, NN);
  runGemm(Gt1, 32, gate[1].ss, 1, gate[2], Gt2, NN);
  statsOf(gate[2], Gt2, NN);
  fill(gmax, NG, -INFINITY);
  fill(wsum, NG, 0.f);
  fill(pooled, NG * 192, 0.f);
  k_gate<<<(NN + 255) / 256, 256, 0, stream>>>(Gt2, 32, gate[2].ss, batch, gbuf, gmax, NN);
  k_poolw<<<(NN + 255) / 256, 256, 0, stream>>>(gbuf, batch, gmax, wbuf, wsum, NN);
  k_poolacc<<<(NN * 192 + 255) / 256, 256, 0, stream>>>(H, wbuf, wsum, batch, pooled, NN);
  k_to_half<<<(NG * 192 + 255) / 256, 256, 0, stream>>>(pooled, NG, 192, 192, Ph);

  // ---- final MLP [192 -> 8] -------------------------------------------------
  runGemm(Ph, 192, unitSS, 0, mlpl, Ym, NG);
  statsOf(mlpl, Ym, NG);
  k_out<<<(NG * LAT + 255) / 256, 256, 0, stream>>>(Ym, 32, mlpl.ss, (float*)d_out, NG, LAT);
}
